// AttentionLayer_5360119185640
// MI455X (gfx1250) — matrix operand order repaired and checked by reading, not compiled, not run on hardware
//
#include <hip/hip_runtime.h>
#include <hip/hip_bf16.h>

#define N_NODES  50000
#define N_EDGES  800000
#define EMBED    128
#define HEADS    8
#define HEAD_DIM 16
#define HIDDEN   512
#define EPS      1e-5f
#define SLOPE    0.2f

typedef __attribute__((ext_vector_type(16))) _Float16 v16h;
typedef __attribute__((ext_vector_type(8)))  _Float16 v8h;
typedef __attribute__((ext_vector_type(8)))  float    v8f;
typedef __attribute__((ext_vector_type(4)))  int      v4i;

// ---------------- CDNA5 async global->LDS staging ----------------

#if defined(__has_builtin) && __has_builtin(__builtin_amdgcn_global_load_async_to_lds_b128)
#define HAVE_ASYNC_LDS 1
#else
#define HAVE_ASYNC_LDS 0
#endif

// Copy 16 bytes (8 halves) global -> LDS, asynchronously if the HW path exists.
__device__ __forceinline__ void async_copy16(const _Float16* g, _Float16* l) {
#if HAVE_ASYNC_LDS
  __builtin_amdgcn_global_load_async_to_lds_b128(
      (__attribute__((address_space(1))) v4i*)g,
      (__attribute__((address_space(3))) v4i*)l, 0, 0);
#else
  *(v8h*)l = *(const v8h*)g;
#endif
}

__device__ __forceinline__ void async_wait() {
#if HAVE_ASYNC_LDS
#if __has_builtin(__builtin_amdgcn_s_wait_asynccnt)
  __builtin_amdgcn_s_wait_asynccnt(0);
#else
  asm volatile("s_wait_asynccnt 0x0" ::: "memory");
#endif
#endif
}

// ---------------- WMMA fragment helpers (CDNA5 16x16x32 f16 layouts) ----------------

__device__ __forceinline__ v16h make_frag(v8h lo, v8h hi) {
  v16h r;
#pragma unroll
  for (int i = 0; i < 8; ++i) { r[i] = lo[i]; r[i + 8] = hi[i]; }
  return r;
}

// A fragment: 16x32 f16, row-major source (leading dim ld, in halves).
// Lane &15 = row; halves j=0..7 -> k=base+j, j=8..15 -> k=base+16+(j-8); base=+8 for lanes>=16.
__device__ __forceinline__ v16h load_a(const _Float16* base, int ld, int lane, int kb) {
  const int row = lane & 15;
  const int k0  = kb + ((lane >> 4) << 3);
  const _Float16* p = base + row * ld + k0;
  v8h lo = *(const v8h*)(p);
  v8h hi = *(const v8h*)(p + 16);
  return make_frag(lo, hi);
}

// B fragment: 32x16 f16 from TRANSPOSED weights bt[col][k] (leading dim ld).
// Lane &15 = col; halves j=0..7 -> k=base+j, j=8..15 -> k=base+16+(j-8); base=kb+8 for lanes>=16.
__device__ __forceinline__ v16h load_b(const _Float16* bt, int ld, int lane, int kb) {
  const int col = lane & 15;
  const int k0  = kb + ((lane >> 4) << 3);
  const _Float16* p = bt + col * ld + k0;
  v8h lo = *(const v8h*)(p);
  v8h hi = *(const v8h*)(p + 16);
  return make_frag(lo, hi);
}

__device__ __forceinline__ v8f wmma_f16(v16h a, v16h b, v8f c) {
  return __builtin_amdgcn_wmma_f32_16x16x32_f16(false, a, false, b, (short)0, c,
                                                false, false);
}

// ---------------- pack / init kernels ----------------

// w: [K][M] row-major f32  ->  wt: [M][K] f16 (transposed, B-friendly)
__global__ __launch_bounds__(256) void k_pack_w(const float* __restrict__ w,
                                                _Float16* __restrict__ wt,
                                                int K, int M) {
  int idx = blockIdx.x * 256 + threadIdx.x;
  if (idx >= K * M) return;
  int k = idx % K, c = idx / K;
  wt[c * K + k] = (_Float16)w[k * M + c];
}

// xh = f16(x); hbuf = x + gat_bias (skip-accumulator initialised with residual+bias)
__global__ __launch_bounds__(256) void k_pack_x(const float* __restrict__ x,
                                                const float* __restrict__ gat_bias,
                                                _Float16* __restrict__ xh,
                                                float* __restrict__ hbuf) {
  int idx = blockIdx.x * 256 + threadIdx.x;
  if (idx >= N_NODES * EMBED) return;
  float v = x[idx];
  xh[idx]   = (_Float16)v;
  hbuf[idx] = v + gat_bias[idx & (EMBED - 1)];
}

__global__ __launch_bounds__(256) void k_init(float* __restrict__ m,
                                              float* __restrict__ denom,
                                              float* __restrict__ stats) {
  int idx = blockIdx.x * 256 + threadIdx.x;
  if (idx < N_NODES * HEADS) {
    m[idx]     = __int_as_float(0xff800000); // -inf
    denom[idx] = 0.f;
  }
  if (idx < 4 * EMBED) stats[idx] = 0.f;
}

// ---------------- feat = x @ W  (WMMA, A staged once per block via async->LDS) -------

__global__ __launch_bounds__(256) void k_gemm_feat(const _Float16* __restrict__ xh,
                                                   const _Float16* __restrict__ wt,
                                                   float* __restrict__ feat) {
  __shared__ _Float16 atile[16 * EMBED];  // 4 KB, shared by all 8 waves
  const int rowTile = blockIdx.x * 16;
  // 16 rows x 128 halves = 4096 B = 256 x b128: exactly one async copy per thread
  async_copy16(xh + rowTile * EMBED + threadIdx.x * 8, atile + threadIdx.x * 8);
  async_wait();
  __syncthreads();

  const int lane = threadIdx.x & 31;
  const int wave = threadIdx.x >> 5;
  const int colTile = wave * 16; // 8 waves cover 128 output cols
  const _Float16* b = wt + colTile * EMBED;
  v8f acc = {};
#pragma unroll
  for (int kb = 0; kb < EMBED; kb += 32)
    acc = wmma_f16(load_a(atile, EMBED, lane, kb), load_b(b, EMBED, lane, kb), acc);
  const int col   = colTile + (lane & 15);
  const int rbase = rowTile + ((lane >> 4) << 3);
#pragma unroll
  for (int i = 0; i < 8; ++i)
    feat[(rbase + i) * EMBED + col] = acc[i];
}

// ---------------- GAT edge softmax + scatter ----------------

__global__ __launch_bounds__(256) void k_el_er(const float* __restrict__ feat,
                                               const float* __restrict__ attn_l,
                                               const float* __restrict__ attn_r,
                                               float* __restrict__ el,
                                               float* __restrict__ er) {
  int idx = blockIdx.x * 256 + threadIdx.x; // node*HEADS + h
  if (idx >= N_NODES * HEADS) return;
  int h = idx & (HEADS - 1), node = idx >> 3;
  const float* f  = feat + node * EMBED + h * HEAD_DIM;
  const float* al = attn_l + h * HEAD_DIM;
  const float* ar = attn_r + h * HEAD_DIM;
  float sl = 0.f, sr = 0.f;
#pragma unroll
  for (int d = 0; d < HEAD_DIM; ++d) { sl += f[d] * al[d]; sr += f[d] * ar[d]; }
  el[idx] = sl; er[idx] = sr;
}

__device__ __forceinline__ void atomicMaxF(float* addr, float val) {
  if (val >= 0.f) atomicMax((int*)addr, __float_as_int(val));
  else            atomicMin((unsigned int*)addr, __float_as_uint(val));
}

__global__ __launch_bounds__(256) void k_edge_max(const int* __restrict__ src,
                                                  const int* __restrict__ dst,
                                                  const float* __restrict__ el,
                                                  const float* __restrict__ er,
                                                  float* __restrict__ ex,
                                                  float* __restrict__ m) {
  int idx = blockIdx.x * 256 + threadIdx.x; // edge*HEADS + h
  if (idx >= N_EDGES * HEADS) return;
  int h = idx & (HEADS - 1), e = idx >> 3;
  int s = src[e], d = dst[e];
  float v = el[s * HEADS + h] + er[d * HEADS + h];
  v = v > 0.f ? v : SLOPE * v;          // leaky relu
  ex[idx] = v;
  atomicMaxF(&m[d * HEADS + h], v);
}

__global__ __launch_bounds__(256) void k_edge_exp(const int* __restrict__ dst,
                                                  const float* __restrict__ m,
                                                  float* __restrict__ ex,
                                                  float* __restrict__ denom) {
  int idx = blockIdx.x * 256 + threadIdx.x;
  if (idx >= N_EDGES * HEADS) return;
  int h = idx & (HEADS - 1), e = idx >> 3;
  int d = dst[e];
  float v = __expf(ex[idx] - m[d * HEADS + h]);
  ex[idx] = v;
  atomicAdd(&denom[d * HEADS + h], v);
}

__global__ __launch_bounds__(256) void k_edge_scatter(const int* __restrict__ src,
                                                      const int* __restrict__ dst,
                                                      const float* __restrict__ ex,
                                                      const float* __restrict__ denom,
                                                      const float* __restrict__ feat,
                                                      float* __restrict__ hbuf) {
  int idx = blockIdx.x * 256 + threadIdx.x;
  if (idx >= N_EDGES * HEADS) return;
  int h = idx & (HEADS - 1), e = idx >> 3;
  int s = src[e], d = dst[e];
  float att = ex[idx] / denom[d * HEADS + h];
  const float4* fs = (const float4*)(feat + s * EMBED + h * HEAD_DIM);
  float* out = hbuf + d * EMBED + h * HEAD_DIM;
#pragma unroll
  for (int q = 0; q < 4; ++q) {
    float4 f = fs[q];
    atomicAdd(out + q * 4 + 0, f.x * att);
    atomicAdd(out + q * 4 + 1, f.y * att);
    atomicAdd(out + q * 4 + 2, f.z * att);
    atomicAdd(out + q * 4 + 3, f.w * att);
  }
}

// ---------------- BatchNorm ----------------

__global__ __launch_bounds__(256) void k_bn_stats(const float* __restrict__ h,
                                                  float* __restrict__ stats) {
  __shared__ float s[EMBED], q[EMBED];
  const int c    = threadIdx.x & (EMBED - 1);
  const int half = threadIdx.x >> 7;
  if (threadIdx.x < EMBED) { s[threadIdx.x] = 0.f; q[threadIdx.x] = 0.f; }
  __syncthreads();
  float ls = 0.f, lq = 0.f;
#pragma unroll 4
  for (int i = 0; i < 64; ++i) {
    int row = blockIdx.x * 128 + half + i * 2;
    if (row < N_NODES) { float v = h[row * EMBED + c]; ls += v; lq += v * v; }
  }
  atomicAdd(&s[c], ls); atomicAdd(&q[c], lq);
  __syncthreads();
  if (threadIdx.x < EMBED) {
    atomicAdd(&stats[threadIdx.x], s[threadIdx.x]);
    atomicAdd(&stats[EMBED + threadIdx.x], q[threadIdx.x]);
  }
}

// Apply BN1: h (in place, f32) and f16 copy for the FFN GEMM.
__global__ __launch_bounds__(256) void k_bn1_apply(float* __restrict__ h,
                                                   _Float16* __restrict__ hbn16,
                                                   const float* __restrict__ stats,
                                                   const float* __restrict__ gamma,
                                                   const float* __restrict__ beta) {
  int idx = blockIdx.x * 256 + threadIdx.x;
  if (idx >= N_NODES * EMBED) return;
  int c = idx & (EMBED - 1);
  const float inv_n = 1.f / (float)N_NODES;
  float mu  = stats[c] * inv_n;
  float var = stats[EMBED + c] * inv_n - mu * mu;
  float rs  = rsqrtf(var + EPS);
  float v   = gamma[c] * (h[idx] - mu) * rs + beta[c];
  h[idx]     = v;
  hbn16[idx] = (_Float16)v;
}

// ---------------- fused FFN: GEMM1 -> ReLU -> LDS -> GEMM2 -> skip -> BN2 stats ------

__global__ __launch_bounds__(256) void k_ffn(const _Float16* __restrict__ hbn16,
                                             const _Float16* __restrict__ w1t,
                                             const _Float16* __restrict__ w2t,
                                             const float* __restrict__ b1,
                                             const float* __restrict__ b2,
                                             const float* __restrict__ hbn,
                                             float* __restrict__ z,
                                             float* __restrict__ stats2) {
  __shared__ _Float16 atile[16 * EMBED];  // 4 KB staged A, reused 32x
  __shared__ _Float16 act[16 * HIDDEN];   // 16 KB ReLU activations, never hit HBM
  __shared__ float s2[EMBED], q2[EMBED];
  const int lane = threadIdx.x & 31;
  const int wave = threadIdx.x >> 5;
  if (threadIdx.x < EMBED) { s2[threadIdx.x] = 0.f; q2[threadIdx.x] = 0.f; }
  const int rowTile = blockIdx.x * 16;

  // stage A (16 x 128 halves = 4 KB): one async b128 per thread
  async_copy16(hbn16 + rowTile * EMBED + threadIdx.x * 8, atile + threadIdx.x * 8);
  async_wait();
  __syncthreads();

  // GEMM1: 16 x 512, 32 column tiles, 4 per wave, K = 128, A from LDS
#pragma unroll
  for (int t = 0; t < 4; ++t) {
    const int colTile = (wave + t * 8) * 16;
    const _Float16* b = w1t + colTile * EMBED;
    v8f acc = {};
#pragma unroll
    for (int kb = 0; kb < EMBED; kb += 32)
      acc = wmma_f16(load_a(atile, EMBED, lane, kb), load_b(b, EMBED, lane, kb), acc);
    const int col  = colTile + (lane & 15);
    const float bv = b1[col];
    const int r0   = (lane >> 4) << 3;
#pragma unroll
    for (int i = 0; i < 8; ++i) {
      float v = acc[i] + bv;
      v = v > 0.f ? v : 0.f; // ReLU
      act[(r0 + i) * HIDDEN + col] = (_Float16)v;
    }
  }
  __syncthreads();

  // GEMM2: 16 x 128 from LDS, 1 column tile per wave, K = 512
  {
    const int colTile = wave * 16;
    const _Float16* b = w2t + colTile * HIDDEN;
    v8f acc = {};
#pragma unroll
    for (int kb = 0; kb < HIDDEN; kb += 32)
      acc = wmma_f16(load_a(act, HIDDEN, lane, kb), load_b(b, HIDDEN, lane, kb), acc);
    const int col  = colTile + (lane & 15);
    const float bv = b2[col];
    const int r0   = (lane >> 4) << 3;
    float ls = 0.f, lq = 0.f;
#pragma unroll
    for (int i = 0; i < 8; ++i) {
      int row = rowTile + r0 + i;
      float v = acc[i] + bv + hbn[row * EMBED + col]; // y + skip(h_bn)
      z[row * EMBED + col] = v;
      ls += v; lq += v * v;
    }
    atomicAdd(&s2[col], ls);
    atomicAdd(&q2[col], lq);
  }
  __syncthreads();
  if (threadIdx.x < EMBED) {
    atomicAdd(&stats2[threadIdx.x], s2[threadIdx.x]);
    atomicAdd(&stats2[EMBED + threadIdx.x], q2[threadIdx.x]);
  }
}

__global__ __launch_bounds__(256) void k_bn2_apply(const float* __restrict__ z,
                                                   const float* __restrict__ stats2,
                                                   const float* __restrict__ gamma,
                                                   const float* __restrict__ beta,
                                                   float* __restrict__ out) {
  int idx = blockIdx.x * 256 + threadIdx.x;
  if (idx >= N_NODES * EMBED) return;
  int c = idx & (EMBED - 1);
  const float inv_n = 1.f / (float)N_NODES;
  float mu  = stats2[c] * inv_n;
  float var = stats2[EMBED + c] * inv_n - mu * mu;
  float rs  = rsqrtf(var + EPS);
  out[idx]  = gamma[c] * (z[idx] - mu) * rs + beta[c];
}

// ---------------- launcher ----------------

extern "C" void kernel_launch(void* const* d_in, const int* in_sizes, int n_in,
                              void* d_out, int out_size, void* d_ws, size_t ws_size,
                              hipStream_t stream) {
  const float* x        = (const float*)d_in[0];
  const int*   src      = (const int*)  d_in[1];
  const int*   dst      = (const int*)  d_in[2];
  const float* W        = (const float*)d_in[3];
  const float* attn_l   = (const float*)d_in[4];
  const float* attn_r   = (const float*)d_in[5];
  const float* gat_bias = (const float*)d_in[6];
  const float* gamma1   = (const float*)d_in[7];
  const float* beta1    = (const float*)d_in[8];
  const float* gamma2   = (const float*)d_in[9];
  const float* beta2    = (const float*)d_in[10];
  const float* W1       = (const float*)d_in[11];
  const float* b1       = (const float*)d_in[12];
  const float* W2       = (const float*)d_in[13];
  const float* b2       = (const float*)d_in[14];
  float* out = (float*)d_out;

  char* ws = (char*)d_ws;
  size_t off = 0;
  auto alloc = [&](size_t bytes) -> void* {
    void* p = ws + off;
    off = (off + bytes + 255) & ~(size_t)255;
    return p;
  };
  const size_t NF = (size_t)N_NODES * EMBED;
  const size_t NH = (size_t)N_NODES * HEADS;
  const size_t EH = (size_t)N_EDGES * HEADS;

  float*    feat  = (float*)   alloc(NF * 4);
  float*    hbuf  = (float*)   alloc(NF * 4);
  float*    zbuf  = (float*)   alloc(NF * 4);
  float*    exbuf = (float*)   alloc(EH * 4);
  _Float16* xh    = (_Float16*)alloc(NF * 2);
  _Float16* hbn16 = (_Float16*)alloc(NF * 2);
  float*    el    = (float*)   alloc(NH * 4);
  float*    er    = (float*)   alloc(NH * 4);
  float*    mmax  = (float*)   alloc(NH * 4);
  float*    denom = (float*)   alloc(NH * 4);
  float*    stats = (float*)   alloc(4 * EMBED * 4); // [BN1 sum|sq, BN2 sum|sq]
  _Float16* wt    = (_Float16*)alloc((size_t)EMBED * EMBED * 2);
  _Float16* w1t   = (_Float16*)alloc((size_t)EMBED * HIDDEN * 2);
  _Float16* w2t   = (_Float16*)alloc((size_t)HIDDEN * EMBED * 2);
  float* stats1 = stats;
  float* stats2 = stats + 2 * EMBED;

  const int gNF = (int)((NF + 255) / 256);   // 25000
  const int gNH = (int)((NH + 255) / 256);   // 1563
  const int gEH = (int)((EH + 255) / 256);   // 25000
  const int gRT = N_NODES / 16;              // 3125 row tiles

  k_pack_w<<<(EMBED * EMBED + 255) / 256, 256, 0, stream>>>(W,  wt,  EMBED,  EMBED);
  k_pack_w<<<(EMBED * HIDDEN + 255) / 256, 256, 0, stream>>>(W1, w1t, EMBED,  HIDDEN);
  k_pack_w<<<(HIDDEN * EMBED + 255) / 256, 256, 0, stream>>>(W2, w2t, HIDDEN, EMBED);
  k_pack_x<<<gNF, 256, 0, stream>>>(x, gat_bias, xh, hbuf);
  k_init  <<<gNH, 256, 0, stream>>>(mmax, denom, stats);

  k_gemm_feat<<<gRT, 256, 0, stream>>>(xh, wt, feat);
  k_el_er    <<<gNH, 256, 0, stream>>>(feat, attn_l, attn_r, el, er);

  k_edge_max    <<<gEH, 256, 0, stream>>>(src, dst, el, er, exbuf, mmax);
  k_edge_exp    <<<gEH, 256, 0, stream>>>(dst, mmax, exbuf, denom);
  k_edge_scatter<<<gEH, 256, 0, stream>>>(src, dst, exbuf, denom, feat, hbuf);

  k_bn_stats <<<(N_NODES + 127) / 128, 256, 0, stream>>>(hbuf, stats1);
  k_bn1_apply<<<gNF, 256, 0, stream>>>(hbuf, hbn16, stats1, gamma1, beta1);

  k_ffn<<<gRT, 256, 0, stream>>>(hbn16, w1t, w2t, b1, b2, hbuf, zbuf, stats2);

  k_bn2_apply<<<gNF, 256, 0, stream>>>(zbuf, stats2, gamma2, beta2, out);
}
